// LambdaLayer_1228360647408
// MI455X (gfx1250) — matrix operand order repaired and checked by reading, not compiled, not run on hardware
//
#include <hip/hip_runtime.h>
#include <hip/hip_bf16.h>

// ---------------------------------------------------------------------------
// LambdaLayer on gfx1250 (MI455X): bf16 WMMA pipeline
//   K0: zero lambda_c accumulator
//   Kp: pack [Wq|Wk|Wv] into WMMA-B fragment order (bf16)
//   K1: X·W + BN + softmax  — LDS-staged A, fragment-ordered B (wide loads)
//   K2: lambda_c = softmax(k)^T · v — LDS-staged, WMMA + global f32 atomics
//   K3: 7x7 positional conv as im2col WMMA in LDS (async-to-LDS staging)
//       + fused Y = q·(lambda_c + lam_p) epilogue
// ---------------------------------------------------------------------------

typedef __attribute__((ext_vector_type(16))) __bf16 v16bf;
typedef __attribute__((ext_vector_type(8)))  float  v8f;
typedef int v4i __attribute__((vector_size(16)));

#define AS1_V4I(p) ((__attribute__((address_space(1))) v4i*)(p))
#define AS3_V4I(p) ((__attribute__((address_space(3))) v4i*)(p))

#if defined(__has_builtin)
#  if __has_builtin(__builtin_amdgcn_global_load_async_to_lds_b128) && \
      __has_builtin(__builtin_amdgcn_s_wait_asynccnt)
#    define USE_ASYNC_LDS 1
#  endif
#endif
#ifndef USE_ASYNC_LDS
#  define USE_ASYNC_LDS 0
#endif

#define B_   16
#define HH_  64
#define WW_  64
#define C_   256
#define NTOT 65536   // B_*HH_*WW_
#define NPB  4096    // HH_*WW_

__device__ __forceinline__ unsigned short f2bf(float f) {
  __bf16 h = (__bf16)f;
  return __builtin_bit_cast(unsigned short, h);
}
__device__ __forceinline__ float bf2f(unsigned short u) {
  __bf16 h = __builtin_bit_cast(__bf16, u);
  return (float)h;
}
// A-matrix (16x32 bf16) K index for VGPR j, lane-group g  (ISA 7.12.2 table)
__device__ __forceinline__ int a_kidx(int j, int g) {
  return (j < 4) ? (8 * g + 2 * j) : (16 + 8 * g + 2 * (j - 4));
}
// B-matrix (32x16 bf16) K (row) index for VGPR j, lane-group g
__device__ __forceinline__ int b_kidx(int j, int g) {
  return (j < 4) ? (8 * g + 2 * j) : (16 + 8 * g + 2 * (j - 4));
}

// ---------------------------------------------------------------------------
__global__ __launch_bounds__(256) void zero_kernel(float* p, int n) {
  int i = blockIdx.x * blockDim.x + threadIdx.x;
  if (i < n) p[i] = 0.0f;
}

// ---------------------------------------------------------------------------
// Kp: pack weights into fragment-ordered bf16: Wpack[(t*8+s)*32 + lane][16]
//     t = N-tile (0..3 q, 4 k, 5..8 v), s = K-step (32 channels each).
__global__ __launch_bounds__(256)
void packw_kernel(const float* __restrict__ Wq, const float* __restrict__ Wk,
                  const float* __restrict__ Wv,
                  unsigned short* __restrict__ Wpack) {
  int id = blockIdx.x * 256 + threadIdx.x;  // 72 frags * 32 lanes = 2304
  if (id >= 72 * 32) return;
  int lane = id & 31, f = id >> 5;
  int t = f >> 3, s = f & 7;
  int g = lane >> 4, nn = lane & 15;
  const float* W;
  int ldw, cb;
  if (t < 4)       { W = Wq; ldw = 64; cb = t * 16; }
  else if (t == 4) { W = Wk; ldw = 16; cb = 0; }
  else             { W = Wv; ldw = 64; cb = (t - 5) * 16; }
  unsigned short* dst = Wpack + (size_t)id * 16;
#pragma unroll
  for (int j = 0; j < 8; ++j) {
    int r0 = s * 32 + b_kidx(j, g);
    dst[2 * j]     = f2bf(W[(size_t)r0 * ldw + cb + nn]);
    dst[2 * j + 1] = f2bf(W[(size_t)(r0 + 1) * ldw + cb + nn]);
  }
}

// ---------------------------------------------------------------------------
// K1: projections. Block = 256 threads = 8 waves, 128 positions.
// x tile staged to LDS as bf16 with float4 global loads; A fragments built
// from LDS with aligned 4B reads (K pairs are contiguous); B fragments are
// single 32B loads from the fragment-ordered Wpack.
__global__ __launch_bounds__(256)
void proj_kernel(const float* __restrict__ x,
                 const unsigned short* __restrict__ Wpack,
                 const float* __restrict__ qg, const float* __restrict__ qb,
                 const float* __restrict__ qm, const float* __restrict__ qvv,
                 const float* __restrict__ vg, const float* __restrict__ vb,
                 const float* __restrict__ vm, const float* __restrict__ vvv,
                 unsigned short* __restrict__ q_s,
                 unsigned short* __restrict__ v_s,
                 unsigned short* __restrict__ ksm) {
  __shared__ alignas(16) unsigned short xs[128][C_];  // 64 KB bf16 x tile

  const int tid = threadIdx.x;
  const size_t base = (size_t)blockIdx.x * 128;

  if (blockIdx.x + 1 < gridDim.x)
    __builtin_prefetch(x + (base + 128) * C_, 0, 0);

  // ---- stage x tile: 128 x 256 f32 -> bf16 LDS ----
  const float4* xv = (const float4*)(x + base * C_);
  for (int u = tid; u < 128 * 64; u += 256) {  // 64 float4 per row
    float4 f = xv[u];
    union { unsigned short us[4]; uint2 u2; } pk;
    pk.us[0] = f2bf(f.x); pk.us[1] = f2bf(f.y);
    pk.us[2] = f2bf(f.z); pk.us[3] = f2bf(f.w);
    int pos = u >> 6;
    int c4  = (u & 63) << 2;
    *(uint2*)&xs[pos][c4] = pk.u2;
  }
  __syncthreads();

  const int lane = tid & 31;
  const int wave = tid >> 5;
  const int g    = lane >> 4;
  const int m    = lane & 15;          // A row / C row-group
  const int nn   = lane & 15;          // C column
  const int prow = wave * 16 + m;
  const int pbase = (int)base + wave * 16;

  // ---- A fragments from LDS (pairs are K-contiguous -> 4B reads) ----
  v16bf a[8];
#pragma unroll
  for (int s = 0; s < 8; ++s) {
#pragma unroll
    for (int j = 0; j < 8; ++j) {
      int k0 = s * 32 + a_kidx(j, g);                     // even -> 4B aligned
      unsigned int w = *(const unsigned int*)&xs[prow][k0];
      a[s][2 * j]     = __builtin_bit_cast(__bf16, (unsigned short)(w & 0xffffu));
      a[s][2 * j + 1] = __builtin_bit_cast(__bf16, (unsigned short)(w >> 16));
    }
  }

  // ---- 9 N-tiles: 0..3 = q, 4 = k, 5..8 = v ----
  for (int t = 0; t < 9; ++t) {
    v8f c = {};
#pragma unroll
    for (int s = 0; s < 8; ++s) {
      const v16bf bfrag =
          *(const v16bf*)(Wpack + (((size_t)(t * 8 + s) * 32) + lane) * 16);
      c = __builtin_amdgcn_wmma_f32_16x16x32_bf16(false, a[s], false, bfrag,
                                                  (short)0, c, false, false);
    }

    if (t == 4) {
      // softmax over the 16 k-channels (N dim) per position (row)
#pragma unroll
      for (int r = 0; r < 8; ++r) {
        float v = c[r];
        float mx = v;
        for (int msk = 1; msk < 16; msk <<= 1)
          mx = fmaxf(mx, __shfl_xor(mx, msk, 32));
        float e = __expf(v - mx);
        float sum = e;
        for (int msk = 1; msk < 16; msk <<= 1)
          sum += __shfl_xor(sum, msk, 32);
        float sm = e / sum;
        int pos = pbase + r + 8 * g;
        ksm[(size_t)pos * 16 + nn] = f2bf(sm);
      }
    } else if (t < 4) {
      int ch = t * 16 + nn;
      float scale = qg[ch] * rsqrtf(qvv[ch] + 1e-3f);
      float mean = qm[ch], beta = qb[ch];
#pragma unroll
      for (int r = 0; r < 8; ++r) {
        float y = (c[r] - mean) * scale + beta;
        int pos = pbase + r + 8 * g;
        q_s[(size_t)pos * 64 + ch] = f2bf(y);
      }
    } else {
      int ch = (t - 5) * 16 + nn;
      float scale = vg[ch] * rsqrtf(vvv[ch] + 1e-3f);
      float mean = vm[ch], beta = vb[ch];
#pragma unroll
      for (int r = 0; r < 8; ++r) {
        float y = (c[r] - mean) * scale + beta;
        int pos = pbase + r + 8 * g;
        v_s[(size_t)pos * 64 + ch] = f2bf(y);
      }
    }
  }
}

// ---------------------------------------------------------------------------
// K2: lambda_c[b,k,v] = sum_n ksm[b,n,k] * v_s[b,n,v]. One wave per
// (batch, 128-position chunk), chunk staged to LDS with uint4 copies;
// split-K accumulation via f32 global atomics.
__global__ __launch_bounds__(32)
void lambdac_kernel(const unsigned short* __restrict__ ksm,
                    const unsigned short* __restrict__ v_s,
                    float* __restrict__ Lc) {
  __shared__ alignas(16) unsigned short ks[128][16];  // 4 KB
  __shared__ alignas(16) unsigned short vs[128][64];  // 16 KB

  const int lane = threadIdx.x;
  const int g  = lane >> 4;
  const int m  = lane & 15;   // A row = k-channel
  const int nn = lane & 15;   // C column (within v-tile)
  const int bb    = blockIdx.x >> 5;
  const int chunk = blockIdx.x & 31;
  const size_t n0 = (size_t)bb * NPB + (size_t)chunk * 128;

  {  // stage chunk to LDS with 16B copies
    const uint4* s0 = (const uint4*)(ksm + n0 * 16);
    uint4* d0 = (uint4*)ks;
    for (int u = lane; u < 256; u += 32) d0[u] = s0[u];
    const uint4* s1 = (const uint4*)(v_s + n0 * 64);
    uint4* d1 = (uint4*)vs;
    for (int u = lane; u < 1024; u += 32) d1[u] = s1[u];
  }
  __syncthreads();

  v8f c[4] = {v8f{}, v8f{}, v8f{}, v8f{}};
#pragma unroll
  for (int s = 0; s < 4; ++s) {
    // A = ksm^T : element (m = k-ch, kk = position)
    v16bf af;
#pragma unroll
    for (int j = 0; j < 8; ++j) {
      int p0 = s * 32 + a_kidx(j, g);
      af[2 * j]     = __builtin_bit_cast(__bf16, ks[p0][m]);
      af[2 * j + 1] = __builtin_bit_cast(__bf16, ks[p0 + 1][m]);
    }
#pragma unroll
    for (int vt = 0; vt < 4; ++vt) {
      v16bf bf;
#pragma unroll
      for (int j = 0; j < 8; ++j) {
        int p0 = s * 32 + b_kidx(j, g);
        bf[2 * j]     = __builtin_bit_cast(__bf16, vs[p0][vt * 16 + nn]);
        bf[2 * j + 1] = __builtin_bit_cast(__bf16, vs[p0 + 1][vt * 16 + nn]);
      }
      c[vt] = __builtin_amdgcn_wmma_f32_16x16x32_bf16(false, af, false, bf,
                                                      (short)0, c[vt], false, false);
    }
  }
#pragma unroll
  for (int vt = 0; vt < 4; ++vt) {
#pragma unroll
    for (int r = 0; r < 8; ++r) {
      int kch = r + 8 * g;
      unsafeAtomicAdd(&Lc[(size_t)bb * 1024 + kch * 64 + vt * 16 + nn], c[vt][r]);
    }
  }
}

// ---------------------------------------------------------------------------
// K3: positional conv as im2col WMMA + fused Y = q·(lambda_c + lam_p).
// One block per (batch, row y, 16-wide x tile). The v-neighborhood is staged
// to LDS via CDNA5 async-to-LDS (if available), 16B per transfer.
__global__ __launch_bounds__(256)
void final_kernel(const unsigned short* __restrict__ q_s,
                  const unsigned short* __restrict__ v_s,
                  const float* __restrict__ Lc,
                  const float* __restrict__ posk,
                  const float* __restrict__ posb,
                  float* __restrict__ out) {
  __shared__ alignas(16) unsigned short nb[7][22][64];     // v-neighborhood
  __shared__ alignas(16) unsigned short Pl[64][16];        // taps, rows>=49 zero
  __shared__ alignas(16) unsigned short lamp[16][64][16];  // lam_p [pos][v][k]
  __shared__ alignas(16) unsigned short qt[16][64];        // q tile
  __shared__ alignas(16) unsigned short lcb[16][64];       // lambda_c bf16 [k][v]

  const int tid = threadIdx.x;
  const int bi  = blockIdx.x;
  const int xt  = bi & 3;
  const int y   = (bi >> 2) & 63;
  const int bb  = bi >> 8;
  const int x0  = xt * 16;

  // ---- zero-fill neighborhood (covers padding), then async-load in-bounds ---
  for (int idx = tid; idx < 7 * 22 * 64; idx += 256)
    (&nb[0][0][0])[idx] = 0;
  // ---- tap matrix (pad 49 -> 64 rows with zero) ----
  for (int idx = tid; idx < 64 * 16; idx += 256) {
    int k = idx & 15, t = idx >> 4;
    Pl[t][k] = f2bf(t < 49 ? posk[t * 16 + k] : 0.0f);
  }
  // ---- q tile ----
  for (int idx = tid; idx < 16 * 64; idx += 256) {
    int ch = idx & 63, p = idx >> 6;
    qt[p][ch] = q_s[((size_t)bb * NPB + y * WW_ + x0 + p) * 64 + ch];
  }
  // ---- lambda_c -> bf16 ----
  for (int idx = tid; idx < 16 * 64; idx += 256) {
    int v = idx & 63, k = idx >> 6;
    lcb[k][v] = f2bf(Lc[(size_t)bb * 1024 + k * 64 + v]);
  }
  __syncthreads();

  // in-bounds neighborhood rows: 7*22 rows, 8 x 16B segments each
  for (int u = tid; u < 7 * 22 * 8; u += 256) {
    int seg = u & 7;
    int rest = u >> 3;
    int xl = rest % 22;
    int dy = rest / 22;
    int yy = y + dy - 3;
    int xx = x0 + xl - 3;
    if (yy >= 0 && yy < HH_ && xx >= 0 && xx < WW_) {
      const unsigned short* src =
          &v_s[((size_t)bb * NPB + yy * WW_ + xx) * 64 + seg * 8];
      unsigned short* dst = &nb[dy][xl][seg * 8];
#if USE_ASYNC_LDS
      __builtin_amdgcn_global_load_async_to_lds_b128(AS1_V4I(src), AS3_V4I(dst),
                                                     0, 0);
#else
      *(uint4*)dst = *(const uint4*)src;
#endif
    }
  }
#if USE_ASYNC_LDS
  __builtin_amdgcn_s_wait_asynccnt(0);
#endif
  __syncthreads();

  const int lane = tid & 31;
  const int wave = tid >> 5;
  const int g  = lane >> 4;
  const int m  = lane & 15;  // position within x tile
  const int nn = lane & 15;  // k-channel (C column)

  // B fragments (shared by all v-channels of this wave)
  v16bf b0, b1;
#pragma unroll
  for (int j = 0; j < 8; ++j) {
    int t0 = b_kidx(j, g);
    b0[2 * j]     = __builtin_bit_cast(__bf16, Pl[t0][nn]);
    b0[2 * j + 1] = __builtin_bit_cast(__bf16, Pl[t0 + 1][nn]);
    int t1 = 32 + t0;
    b1[2 * j]     = __builtin_bit_cast(__bf16, Pl[t1][nn]);
    b1[2 * j + 1] = __builtin_bit_cast(__bf16, Pl[t1 + 1][nn]);
  }
  const float bias = posb[nn];

#pragma unroll
  for (int vi = 0; vi < 8; ++vi) {
    int v = wave * 8 + vi;
    v16bf a0, a1;
#pragma unroll
    for (int j = 0; j < 8; ++j) {
      int t = a_kidx(j, g);  // taps 0..31
      {
        int dy = t / 7, dx = t % 7;
        a0[2 * j] = __builtin_bit_cast(__bf16, nb[dy][m + dx][v]);
        int t2 = t + 1, dy2 = t2 / 7, dx2 = t2 % 7;
        a0[2 * j + 1] = __builtin_bit_cast(__bf16, nb[dy2][m + dx2][v]);
      }
      {
        int t2 = 32 + t;  // taps 32..63; rows >= 49 multiply by zero taps
        int ta = t2 > 48 ? 48 : t2;
        int tb = t2 + 1 > 48 ? 48 : t2 + 1;
        int dya = ta / 7, dxa = ta % 7, dyb = tb / 7, dxb = tb % 7;
        a1[2 * j]     = __builtin_bit_cast(__bf16, nb[dya][m + dxa][v]);
        a1[2 * j + 1] = __builtin_bit_cast(__bf16, nb[dyb][m + dxb][v]);
      }
    }
    v8f c = {};
    c = __builtin_amdgcn_wmma_f32_16x16x32_bf16(false, a0, false, b0,
                                                (short)0, c, false, false);
    c = __builtin_amdgcn_wmma_f32_16x16x32_bf16(false, a1, false, b1,
                                                (short)0, c, false, false);
#pragma unroll
    for (int r = 0; r < 8; ++r) {
      int p = r + 8 * g;
      lamp[p][v][nn] = f2bf(c[r] + bias);
    }
  }
  __syncthreads();

  // ---- fused epilogue: Y[p,h,v] = sum_k q[p,h*16+k]*(Lc[k,v]+lamp[p,v,k]) --
  for (int o = tid; o < 4096; o += 256) {
    int v = o & 63;
    int h = (o >> 6) & 3;
    int p = o >> 8;
    float acc = 0.0f;
#pragma unroll
    for (int k = 0; k < 16; ++k) {
      float qv = bf2f(qt[p][h * 16 + k]);
      float l  = bf2f(lcb[k][v]) + bf2f(lamp[p][v][k]);
      acc += qv * l;
    }
    out[((size_t)bb * NPB + y * WW_ + x0 + p) * 256 + h * 64 + v] = acc;
  }
}

// ---------------------------------------------------------------------------
extern "C" void kernel_launch(void* const* d_in, const int* in_sizes, int n_in,
                              void* d_out, int out_size, void* d_ws,
                              size_t ws_size, hipStream_t stream) {
  const float* x    = (const float*)d_in[0];
  const float* Wq   = (const float*)d_in[1];
  const float* Wk   = (const float*)d_in[2];
  const float* Wv   = (const float*)d_in[3];
  const float* qg   = (const float*)d_in[4];
  const float* qb   = (const float*)d_in[5];
  const float* qm   = (const float*)d_in[6];
  const float* qv   = (const float*)d_in[7];
  const float* vg   = (const float*)d_in[8];
  const float* vb   = (const float*)d_in[9];
  const float* vm   = (const float*)d_in[10];
  const float* vv   = (const float*)d_in[11];
  const float* posk = (const float*)d_in[12];
  const float* posb = (const float*)d_in[13];
  float* out = (float*)d_out;

  // workspace layout (bf16 stored as u16), all regions 256B-aligned
  unsigned short* q_s = (unsigned short*)d_ws;                 // 65536*64
  unsigned short* v_s = q_s + (size_t)NTOT * 64;               // 65536*64
  unsigned short* ksm = v_s + (size_t)NTOT * 64;               // 65536*16
  float* Lc = (float*)(ksm + (size_t)NTOT * 16);               // 16*16*64 f32
  unsigned short* Wpack = (unsigned short*)(Lc + B_ * 16 * 64);  // 72*32*16

  zero_kernel<<<64, 256, 0, stream>>>(Lc, B_ * 16 * 64);
  packw_kernel<<<9, 256, 0, stream>>>(Wq, Wk, Wv, Wpack);
  proj_kernel<<<NTOT / 128, 256, 0, stream>>>(x, Wpack, qg, qb, qm, qv,
                                              vg, vb, vm, vv, q_s, v_s, ksm);
  lambdac_kernel<<<B_ * 32, 32, 0, stream>>>(ksm, v_s, Lc);
  final_kernel<<<B_ * HH_ * 4, 256, 0, stream>>>(q_s, v_s, Lc, posk, posb, out);
}
